// Memorynet_81990925680879
// MI455X (gfx1250) — matrix operand order repaired and checked by reading, not compiled, not run on hardware
//
#include <hip/hip_runtime.h>
#include <cmath>

// ---------------- problem constants ----------------
#define D_CH    256
#define C_NUM   13
#define C_PAD   16                   // classes padded to one WMMA N-tile
#define MEM_LEN 128
#define D_FF    1024
#define NPOINTS (8 * 16384)          // B*N = 131072
#define TILE_M  64
#define NBLK    (NPOINTS / TILE_M)   // 2048
#define LDA     264                  // 256 + 8 bf16 pad -> conflict-free ds_load_b128
#define LDATT   40                   // attn-weight row stride (bf16), conflict-free
#define EPSN    1e-12f

typedef __bf16 bf16;
typedef __attribute__((ext_vector_type(8)))  __bf16 v8bf;
typedef __attribute__((ext_vector_type(16))) __bf16 v16bf;
typedef __attribute__((ext_vector_type(8)))  float  v8f;

union V16U { v16bf v; v8bf h[2]; };

static __device__ __forceinline__ v16bf cat16(v8bf lo, v8bf hi) {
    V16U u; u.h[0] = lo; u.h[1] = hi; return u.v;
}

// A-matrix 16x32 bf16 fragment set for one 16-row strip, K=256 (8 k-steps).
// ISA 7.12.2: lanes 0-15 row M=lane hold K {0..7,16..23}; lanes 16-31 hold {8..15,24..31}.
struct AStrip { v16bf a[8]; };

static __device__ __forceinline__ void load_astrip(AStrip& s, const bf16* Abase,
                                                   int lidx, int lhalf) {
    const bf16* p = Abase + lidx * LDA + lhalf * 8;
#pragma unroll
    for (int ks = 0; ks < 8; ++ks) {
        v8bf lo = *(const v8bf*)(p + ks * 32);
        v8bf hi = *(const v8bf*)(p + ks * 32 + 16);
        s.a[ks] = cat16(lo, hi);
    }
}

// Single K=32 A fragment (for attn @ v), row stride ldaa.
static __device__ __forceinline__ v16bf load_afrag32(const bf16* Abase, int ldaa,
                                                     int lidx, int lhalf) {
    const bf16* p = Abase + lidx * ldaa + lhalf * 8;
    v8bf lo = *(const v8bf*)p;
    v8bf hi = *(const v8bf*)(p + 16);
    return cat16(lo, hi);
}

// B-matrix 32x16 bf16 fragment: lane (lidx) = column N; lanes 0-15 K {0..7,16..23}; lanes 16-31 K {8..15,24..31}.
// Bcol is column-major: column n contiguous over K with leading dim ldb.
static __device__ __forceinline__ v16bf load_bfrag(const bf16* __restrict__ Bcol, int ldb,
                                                   int col, int kbase, int lhalf) {
    const bf16* p = Bcol + (size_t)col * ldb + kbase + lhalf * 8;
    v8bf lo = *(const v8bf*)p;
    v8bf hi = *(const v8bf*)(p + 16);
    return cat16(lo, hi);
}

// Dual-strip 16x16 tiles over K=256: one B fragment feeds TWO WMMAs (halves B traffic).
static __device__ __forceinline__ void tile_mm2(const AStrip& A0, const AStrip& A1,
                                                const bf16* __restrict__ Bcol, int ldb,
                                                int kbase, int col, int lhalf,
                                                v8f& acc0, v8f& acc1) {
#pragma unroll
    for (int ks = 0; ks < 8; ++ks) {
        v16bf b = load_bfrag(Bcol, ldb, col, kbase + ks * 32, lhalf);
        acc0 = __builtin_amdgcn_wmma_f32_16x16x32_bf16(false, A0.a[ks], false, b,
                                                       (short)0, acc0, false, false);
        acc1 = __builtin_amdgcn_wmma_f32_16x16x32_bf16(false, A1.a[ks], false, b,
                                                       (short)0, acc1, false, false);
    }
}

// Single-strip variant (K3 keeps persistent acc[16]).
static __device__ __forceinline__ v8f tile_mm(const AStrip& s, const bf16* __restrict__ Bcol,
                                              int ldb, int kbase, int col, int lhalf) {
    v8f acc = {0.f, 0.f, 0.f, 0.f, 0.f, 0.f, 0.f, 0.f};
#pragma unroll
    for (int ks = 0; ks < 8; ++ks) {
        v16bf b = load_bfrag(Bcol, ldb, col, kbase + ks * 32, lhalf);
        acc = __builtin_amdgcn_wmma_f32_16x16x32_bf16(false, s.a[ks], false, b,
                                                      (short)0, acc, false, false);
    }
    return acc;
}

// ============ K1: memory summary (bf16 B-operands), loss reset, weight conversion ============
__global__ __launch_bounds__(256) void k1_prep(
    const float* __restrict__ memory,
    const float* __restrict__ Wq, const float* __restrict__ Wk,
    const float* __restrict__ Wv, const float* __restrict__ Wo,
    const float* __restrict__ pw1, const float* __restrict__ pw2, const float* __restrict__ pw3,
    const float* __restrict__ w1, const float* __restrict__ w2,
    bf16* __restrict__ mfb, bf16* __restrict__ kb, bf16* __restrict__ vbT,
    bf16* __restrict__ pw1b, bf16* __restrict__ pw2b, bf16* __restrict__ pw3b,
    bf16* __restrict__ wqtb, bf16* __restrict__ wotb,
    bf16* __restrict__ w1b, bf16* __restrict__ w2b,
    float* __restrict__ loss)
{
    const int b = blockIdx.x, t = threadIdx.x;
    __shared__ float red[256];
    __shared__ float mfs[256];

    if (b < C_NUM) {
        float s = 0.f;
        const float* mp = memory + (size_t)b * MEM_LEN * D_CH + t;
        for (int r = 0; r < MEM_LEN; ++r) s += mp[(size_t)r * D_CH];
        float mean = s * (1.f / MEM_LEN);
        red[t] = mean * mean;
        __syncthreads();
        for (int off = 128; off > 0; off >>= 1) {
            if (t < off) red[t] += red[t + off];
            __syncthreads();
        }
        float inv = 1.f / fmaxf(sqrtf(red[0]), EPSN);
        float m = mean * inv;
        mfb[b * D_CH + t] = (bf16)m;      // B-operand for contrast logits
        mfs[t] = m;
        __syncthreads();
        float ks = 0.f, vs = 0.f;
        for (int d = 0; d < D_CH; ++d) {
            float mm = mfs[d];
            ks += mm * Wk[d * D_CH + t];
            vs += mm * Wv[d * D_CH + t];
        }
        kb[b * D_CH + t] = (bf16)ks;      // B-operand for scores
        vbT[t * 32 + b]  = (bf16)vs;      // B-operand for attn@v: [n*32 + c]
    } else if (b == C_NUM) {
        if (t == 0) *loss = 0.f;
        for (int i = t; i < (C_PAD - C_NUM) * D_CH; i += 256) {
            mfb[C_NUM * D_CH + i] = (bf16)0.f;
            kb[C_NUM * D_CH + i]  = (bf16)0.f;
        }
        for (int i = t; i < D_CH * 32; i += 256) {
            if ((i & 31) >= C_NUM) vbT[i] = (bf16)0.f;
        }
    } else {
        int cb = b - C_NUM - 1;
        const float* src; bf16* dst; int base; int trans = 0;
        if      (cb <  16) { src = pw1; dst = pw1b; base = cb * 4096; }
        else if (cb <  32) { src = pw2; dst = pw2b; base = (cb - 16) * 4096; }
        else if (cb <  48) { src = pw3; dst = pw3b; base = (cb - 32) * 4096; }
        else if (cb <  64) { src = Wq;  dst = wqtb; base = (cb - 48) * 4096; trans = 1; }
        else if (cb <  80) { src = Wo;  dst = wotb; base = (cb - 64) * 4096; trans = 1; }
        else if (cb < 144) { src = w1;  dst = w1b;  base = (cb - 80) * 4096; }
        else               { src = w2;  dst = w2b;  base = (cb - 144) * 4096; }
        for (int j = 0; j < 16; ++j) {
            int i = base + t + j * 256;
            float v = trans ? src[(i & 255) * D_CH + (i >> 8)] : src[i];
            dst[i] = (bf16)v;
        }
    }
}

// ============ K2: fused proj-MLP + contrast + cross-attention + residual ============
// 2 waves, each owning TWO 16-row strips (dual-M reuse of every B fragment).
__global__ __launch_bounds__(64) void k2_attn_proj(
    const float* __restrict__ features, const int* __restrict__ gts,
    const bf16* __restrict__ mfb, const bf16* __restrict__ kb, const bf16* __restrict__ vbT,
    const bf16* __restrict__ pw1b, const float* __restrict__ pg1, const float* __restrict__ pb1,
    const bf16* __restrict__ pw2b, const float* __restrict__ pg2, const float* __restrict__ pb2,
    const bf16* __restrict__ pw3b, const float* __restrict__ pb3,
    const bf16* __restrict__ wqtb, const bf16* __restrict__ wotb,
    float* __restrict__ out, float* __restrict__ loss)
{
    extern __shared__ char smem[];
    bf16* Abuf = (bf16*)smem;                    // 64 x 264
    bf16* Hbuf = Abuf + TILE_M * LDA;
    bf16* Gbuf = Hbuf + TILE_M * LDA;
    float* fsum   = (float*)(Gbuf + TILE_M * LDA);   // 64
    float* snorm  = fsum + TILE_M;               // 64
    float* projss = snorm + TILE_M;              // 64
    float* cms    = projss + TILE_M;             // 64 x 16 contrast logits
    float* scs    = cms + TILE_M * C_PAD;        // 64 x 16 attention scores
    bf16*  attnb  = (bf16*)(scs + TILE_M * C_PAD);   // 64 x 40 (K=32 padded attn weights)

    const int t = threadIdx.x;
    const int lane = t & 31, wave = t >> 5;
    const int lidx = lane & 15, lhalf = lane >> 4;
    const int m0a = wave * 32;          // first strip
    const int m0b = m0a + 16;           // second strip
    const size_t p0 = (size_t)blockIdx.x * TILE_M;
    const v8f Z8 = {0.f, 0.f, 0.f, 0.f, 0.f, 0.f, 0.f, 0.f};

    // phase 1: stage X (f32 -> bf16), one row per thread, + row sumsq
    {
        int row = t;
        const float* src = features + (p0 + row) * (size_t)D_CH;
        bf16* dstr = Abuf + row * LDA;
        float ss = 0.f;
        for (int j = 0; j < D_CH; j += 4) {
            float4 f = *(const float4*)(src + j);
            dstr[j]     = (bf16)f.x; dstr[j + 1] = (bf16)f.y;
            dstr[j + 2] = (bf16)f.z; dstr[j + 3] = (bf16)f.w;
            ss += f.x * f.x + f.y * f.y + f.z * f.z + f.w * f.w;
        }
        fsum[t] = ss;
    }
    __syncthreads();

    // phase 2: feature inv-norms + GEMM1: h1 = relu(X@pw1^T * pg1 + pb1)
    snorm[t] = 1.f / fmaxf(sqrtf(fsum[t]), EPSN);
    {
        AStrip A0, A1;
        load_astrip(A0, Abuf + m0a * LDA, lidx, lhalf);
        load_astrip(A1, Abuf + m0b * LDA, lidx, lhalf);
        for (int nt = 0; nt < 16; ++nt) {
            int col = nt * 16 + lidx;
            if (nt < 15) __builtin_prefetch((const void*)(pw1b + (size_t)(col + 16) * D_CH), 0, 1);
            v8f acc0 = Z8, acc1 = Z8;
            tile_mm2(A0, A1, pw1b, D_CH, 0, col, lhalf, acc0, acc1);
            float g = pg1[col], bb = pb1[col];
#pragma unroll
            for (int r = 0; r < 8; ++r) {
                Hbuf[(m0a + lhalf * 8 + r) * LDA + col] = (bf16)fmaxf(acc0[r] * g + bb, 0.f);
                Hbuf[(m0b + lhalf * 8 + r) * LDA + col] = (bf16)fmaxf(acc1[r] * g + bb, 0.f);
            }
        }
    }
    __syncthreads();

    // phase 3: GEMM2: h2 = relu(h1@pw2^T * pg2 + pb2)
    {
        AStrip A0, A1;
        load_astrip(A0, Hbuf + m0a * LDA, lidx, lhalf);
        load_astrip(A1, Hbuf + m0b * LDA, lidx, lhalf);
        for (int nt = 0; nt < 16; ++nt) {
            int col = nt * 16 + lidx;
            if (nt < 15) __builtin_prefetch((const void*)(pw2b + (size_t)(col + 16) * D_CH), 0, 1);
            v8f acc0 = Z8, acc1 = Z8;
            tile_mm2(A0, A1, pw2b, D_CH, 0, col, lhalf, acc0, acc1);
            float g = pg2[col], bb = pb2[col];
#pragma unroll
            for (int r = 0; r < 8; ++r) {
                Gbuf[(m0a + lhalf * 8 + r) * LDA + col] = (bf16)fmaxf(acc0[r] * g + bb, 0.f);
                Gbuf[(m0b + lhalf * 8 + r) * LDA + col] = (bf16)fmaxf(acc1[r] * g + bb, 0.f);
            }
        }
    }
    __syncthreads();

    // phase 4: GEMM3: proj = h2@pw3^T + pb3
    {
        AStrip A0, A1;
        load_astrip(A0, Gbuf + m0a * LDA, lidx, lhalf);
        load_astrip(A1, Gbuf + m0b * LDA, lidx, lhalf);
        for (int nt = 0; nt < 16; ++nt) {
            int col = nt * 16 + lidx;
            if (nt < 15) __builtin_prefetch((const void*)(pw3b + (size_t)(col + 16) * D_CH), 0, 1);
            v8f acc0 = Z8, acc1 = Z8;
            tile_mm2(A0, A1, pw3b, D_CH, 0, col, lhalf, acc0, acc1);
            float bb = pb3[col];
#pragma unroll
            for (int r = 0; r < 8; ++r) {
                Hbuf[(m0a + lhalf * 8 + r) * LDA + col] = (bf16)(acc0[r] + bb);
                Hbuf[(m0b + lhalf * 8 + r) * LDA + col] = (bf16)(acc1[r] + bb);
            }
        }
    }
    __syncthreads();

    // phase 5: proj row sumsq (vector LDS reads) + GEMM4: q = snorm[m] * (X@Wq)
    {
        const v8bf* pr = (const v8bf*)(Hbuf + t * LDA);
        float ss = 0.f;
        for (int j = 0; j < D_CH / 8; ++j) {
            v8bf v = pr[j];
#pragma unroll
            for (int e = 0; e < 8; ++e) { float f = (float)v[e]; ss += f * f; }
        }
        projss[t] = ss;
    }
    {
        float sr0[8], sr1[8];
#pragma unroll
        for (int r = 0; r < 8; ++r) {
            sr0[r] = snorm[m0a + lhalf * 8 + r];
            sr1[r] = snorm[m0b + lhalf * 8 + r];
        }
        AStrip A0, A1;
        load_astrip(A0, Abuf + m0a * LDA, lidx, lhalf);
        load_astrip(A1, Abuf + m0b * LDA, lidx, lhalf);
        for (int nt = 0; nt < 16; ++nt) {
            int col = nt * 16 + lidx;
            if (nt < 15) __builtin_prefetch((const void*)(wqtb + (size_t)(col + 16) * D_CH), 0, 1);
            v8f acc0 = Z8, acc1 = Z8;
            tile_mm2(A0, A1, wqtb, D_CH, 0, col, lhalf, acc0, acc1);
#pragma unroll
            for (int r = 0; r < 8; ++r) {
                Gbuf[(m0a + lhalf * 8 + r) * LDA + col] = (bf16)(acc0[r] * sr0[r]);
                Gbuf[(m0b + lhalf * 8 + r) * LDA + col] = (bf16)(acc1[r] * sr1[r]);
            }
        }
    }
    __syncthreads();

    // phase 6: WMMA contrast logits (proj @ mem_feat^T) and scores (q @ k^T), N=16 tile
    {
        AStrip A0, A1;
        load_astrip(A0, Hbuf + m0a * LDA, lidx, lhalf);
        load_astrip(A1, Hbuf + m0b * LDA, lidx, lhalf);
        v8f acc0 = Z8, acc1 = Z8;
        tile_mm2(A0, A1, mfb, D_CH, 0, lidx, lhalf, acc0, acc1);
#pragma unroll
        for (int r = 0; r < 8; ++r) {
            cms[(m0a + lhalf * 8 + r) * C_PAD + lidx] = acc0[r];
            cms[(m0b + lhalf * 8 + r) * C_PAD + lidx] = acc1[r];
        }
    }
    {
        AStrip A0, A1;
        load_astrip(A0, Gbuf + m0a * LDA, lidx, lhalf);
        load_astrip(A1, Gbuf + m0b * LDA, lidx, lhalf);
        v8f acc0 = Z8, acc1 = Z8;
        tile_mm2(A0, A1, kb, D_CH, 0, lidx, lhalf, acc0, acc1);
#pragma unroll
        for (int r = 0; r < 8; ++r) {
            scs[(m0a + lhalf * 8 + r) * C_PAD + lidx] = acc0[r];
            scs[(m0b + lhalf * 8 + r) * C_PAD + lidx] = acc1[r];
        }
    }
    __syncthreads();

    // phase 7: per-row 13-way softmaxes; each thread owns one row (both branches)
    {
        int row = t;
        // contrast NLL
        float pn = 1.f / fmaxf(sqrtf(projss[row]), EPSN);
        float lg[C_NUM], mx = -3.4e38f;
#pragma unroll
        for (int c = 0; c < C_NUM; ++c) {
            float l = cms[row * C_PAD + c] * pn;   // tao = 1
            lg[c] = l; mx = fmaxf(mx, l);
        }
        float se = 0.f;
#pragma unroll
        for (int c = 0; c < C_NUM; ++c) se += expf(lg[c] - mx);
        float lse = mx + logf(se);
        int g = gts[p0 + row];
        atomicAdd(loss, (lse - lg[g]) * (1.f / (float)NPOINTS));
        // attention softmax -> bf16 A-operand (K padded to 32 with zeros)
        float sc[C_NUM]; mx = -3.4e38f;
#pragma unroll
        for (int c = 0; c < C_NUM; ++c) {
            float s = scs[row * C_PAD + c] * 0.0625f;  // 1/sqrt(256)
            sc[c] = s; mx = fmaxf(mx, s);
        }
        se = 0.f;
#pragma unroll
        for (int c = 0; c < C_NUM; ++c) { float e = expf(sc[c] - mx); sc[c] = e; se += e; }
        float inv = 1.f / se;
        bf16* ar = attnb + row * LDATT;
#pragma unroll
        for (int c = 0; c < C_NUM; ++c) ar[c] = (bf16)(sc[c] * inv);
#pragma unroll
        for (int c = C_NUM; c < 32; ++c) ar[c] = (bf16)0.f;
    }
    __syncthreads();

    // phase 8: GEMM attn @ v (K=32, zero-padded) -> Abuf (X dead)
    {
        v16bf af0 = load_afrag32(attnb + m0a * LDATT, LDATT, lidx, lhalf);
        v16bf af1 = load_afrag32(attnb + m0b * LDATT, LDATT, lidx, lhalf);
        for (int nt = 0; nt < 16; ++nt) {
            int col = nt * 16 + lidx;
            v16bf b = load_bfrag(vbT, 32, col, 0, lhalf);
            v8f acc0 = __builtin_amdgcn_wmma_f32_16x16x32_bf16(false, af0, false, b,
                                                               (short)0, Z8, false, false);
            v8f acc1 = __builtin_amdgcn_wmma_f32_16x16x32_bf16(false, af1, false, b,
                                                               (short)0, Z8, false, false);
#pragma unroll
            for (int r = 0; r < 8; ++r) {
                Abuf[(m0a + lhalf * 8 + r) * LDA + col] = (bf16)acc0[r];
                Abuf[(m0b + lhalf * 8 + r) * LDA + col] = (bf16)acc1[r];
            }
        }
    }
    __syncthreads();

    // phase 9: GEMM5: res1 = features + a@Wo  -> out (f32)
    {
        AStrip A0, A1;
        load_astrip(A0, Abuf + m0a * LDA, lidx, lhalf);
        load_astrip(A1, Abuf + m0b * LDA, lidx, lhalf);
        for (int nt = 0; nt < 16; ++nt) {
            int col = nt * 16 + lidx;
            if (nt < 15) __builtin_prefetch((const void*)(wotb + (size_t)(col + 16) * D_CH), 0, 1);
            v8f acc0 = Z8, acc1 = Z8;
            tile_mm2(A0, A1, wotb, D_CH, 0, col, lhalf, acc0, acc1);
#pragma unroll
            for (int r = 0; r < 8; ++r) {
                int rowa = m0a + lhalf * 8 + r;
                int rowb = m0b + lhalf * 8 + r;
                size_t ga = (p0 + rowa) * (size_t)D_CH + col;
                size_t gb2 = (p0 + rowb) * (size_t)D_CH + col;
                out[ga]  = features[ga]  + acc0[r];
                out[gb2] = features[gb2] + acc1[r];
            }
        }
    }
}

// ============ K3: FFN 256 -> 1024 -> 256 (+residual, in-place on out) ============
__global__ __launch_bounds__(128) void k3_ffn(
    const bf16* __restrict__ w1b, const float* __restrict__ b1,
    const float* __restrict__ g1, const float* __restrict__ be1,
    const bf16* __restrict__ w2b, const float* __restrict__ b2,
    float* __restrict__ out)
{
    extern __shared__ char smem[];
    bf16* Rbuf = (bf16*)smem;              // 64 x 264 (res1 bf16)
    bf16* Mbuf = Rbuf + TILE_M * LDA;      // 64 x 264 (m chunk)

    const int t = threadIdx.x;
    const int lane = t & 31, wave = t >> 5;
    const int lidx = lane & 15, lhalf = lane >> 4;
    const int m0 = wave * 16;
    const size_t p0 = (size_t)blockIdx.x * TILE_M;

    {
        int row = t >> 1, c0 = (t & 1) * 128;
        const float* src = out + (p0 + row) * (size_t)D_CH + c0;
        bf16* dstr = Rbuf + row * LDA + c0;
        for (int j = 0; j < 128; j += 4) {
            float4 f = *(const float4*)(src + j);
            dstr[j]     = (bf16)f.x; dstr[j + 1] = (bf16)f.y;
            dstr[j + 2] = (bf16)f.z; dstr[j + 3] = (bf16)f.w;
        }
    }
    __syncthreads();

    v8f acc[16];
    const v8f vzero = {0.f, 0.f, 0.f, 0.f, 0.f, 0.f, 0.f, 0.f};
#pragma unroll
    for (int nt = 0; nt < 16; ++nt) acc[nt] = vzero;

    for (int ch = 0; ch < 4; ++ch) {
        // m = relu((res1 @ w1^T + b1) * g1 + be1), 256-wide FF chunk
        {
            AStrip A; load_astrip(A, Rbuf + m0 * LDA, lidx, lhalf);
            for (int nt = 0; nt < 16; ++nt) {
                int col = nt * 16 + lidx;
                int f = ch * 256 + col;
                if (nt < 15) __builtin_prefetch((const void*)(w1b + (size_t)(f + 16) * D_CH), 0, 1);
                v8f m = tile_mm(A, w1b, D_CH, 0, f, lhalf);
                float bb = b1[f], gg = g1[f], ee = be1[f];
#pragma unroll
                for (int r = 0; r < 8; ++r)
                    Mbuf[(m0 + lhalf * 8 + r) * LDA + col] =
                        (bf16)fmaxf((m[r] + bb) * gg + ee, 0.f);
            }
        }
        __syncthreads();
        // acc += m @ w2^T (chunk of K)
        {
            AStrip A; load_astrip(A, Mbuf + m0 * LDA, lidx, lhalf);
#pragma unroll
            for (int nt = 0; nt < 16; ++nt) {
                int col = nt * 16 + lidx;
#pragma unroll
                for (int ks = 0; ks < 8; ++ks) {
                    v16bf b = load_bfrag(w2b, D_FF, col, ch * 256 + ks * 32, lhalf);
                    acc[nt] = __builtin_amdgcn_wmma_f32_16x16x32_bf16(
                        false, A.a[ks], false, b, (short)0, acc[nt], false, false);
                }
            }
        }
        __syncthreads();
    }

    // epilogue: out = res1 + acc + b2
#pragma unroll
    for (int nt = 0; nt < 16; ++nt) {
        int col = nt * 16 + lidx;
        float bb = b2[col];
#pragma unroll
        for (int r = 0; r < 8; ++r) {
            int row = m0 + lhalf * 8 + r;
            size_t gi = (p0 + row) * (size_t)D_CH + col;
            out[gi] = out[gi] + acc[nt][r] + bb;
        }
    }
}

// ============================= host launcher =============================
extern "C" void kernel_launch(void* const* d_in, const int* in_sizes, int n_in,
                              void* d_out, int out_size, void* d_ws, size_t ws_size,
                              hipStream_t stream)
{
    (void)in_sizes; (void)n_in; (void)out_size; (void)ws_size;
    const float* features = (const float*)d_in[0];
    /* d_in[1] coarse_pred is unused by the reference */
    const int*   gts      = (const int*)d_in[2];
    const float* memory   = (const float*)d_in[3];
    const float* Wq  = (const float*)d_in[4];
    const float* Wk  = (const float*)d_in[5];
    const float* Wv  = (const float*)d_in[6];
    const float* Wo  = (const float*)d_in[7];
    const float* w1  = (const float*)d_in[8];
    const float* b1  = (const float*)d_in[9];
    const float* g1  = (const float*)d_in[10];
    const float* be1 = (const float*)d_in[11];
    const float* w2  = (const float*)d_in[12];
    const float* b2  = (const float*)d_in[13];
    const float* pw1 = (const float*)d_in[14];
    const float* pg1 = (const float*)d_in[15];
    const float* pb1 = (const float*)d_in[16];
    const float* pw2 = (const float*)d_in[17];
    const float* pg2 = (const float*)d_in[18];
    const float* pb2 = (const float*)d_in[19];
    const float* pw3 = (const float*)d_in[20];
    const float* pb3 = (const float*)d_in[21];

    float* out  = (float*)d_out;
    float* loss = out + (size_t)NPOINTS * D_CH;

    // workspace carve-up (~1.7 MB, all regions 16B-aligned)
    bf16* mfb  = (bf16*)d_ws;            // 16 x 256 (rows 13..15 zero)
    bf16* kb   = mfb + C_PAD * D_CH;     // 16 x 256 (rows 13..15 zero)
    bf16* vbT  = kb  + C_PAD * D_CH;     // 256 x 32 (cols 13..31 zero)
    bf16* pw1b = vbT + D_CH * 32;
    bf16* pw2b = pw1b + D_CH * D_CH;
    bf16* pw3b = pw2b + D_CH * D_CH;
    bf16* wqtb = pw3b + D_CH * D_CH;
    bf16* wotb = wqtb + D_CH * D_CH;
    bf16* w1b  = wotb + D_CH * D_CH;
    bf16* w2b  = w1b  + D_FF * D_CH;

    const size_t smem2 = (size_t)3 * TILE_M * LDA * sizeof(bf16)
                       + (size_t)(3 * TILE_M + 2 * TILE_M * C_PAD) * sizeof(float)
                       + (size_t)TILE_M * LDATT * sizeof(bf16);
    const size_t smem3 = (size_t)2 * TILE_M * LDA * sizeof(bf16);

    (void)hipFuncSetAttribute((const void*)k2_attn_proj,
                              hipFuncAttributeMaxDynamicSharedMemorySize, (int)smem2);
    (void)hipFuncSetAttribute((const void*)k3_ffn,
                              hipFuncAttributeMaxDynamicSharedMemorySize, (int)smem3);

    k1_prep<<<C_NUM + 1 + 208, 256, 0, stream>>>(
        memory, Wq, Wk, Wv, Wo, pw1, pw2, pw3, w1, w2,
        mfb, kb, vbT, pw1b, pw2b, pw3b, wqtb, wotb, w1b, w2b, loss);

    k2_attn_proj<<<NBLK, 64, smem2, stream>>>(
        features, gts, mfb, kb, vbT,
        pw1b, pg1, pb1, pw2b, pg2, pb2, pw3b, pb3,
        wqtb, wotb, out, loss);

    k3_ffn<<<NBLK, 128, smem3, stream>>>(w1b, b1, g1, be1, w2b, b2, out);
}
